// CMP_32427003085025
// MI455X (gfx1250) — matrix operand order repaired and checked by reading, not compiled, not run on hardware
//
#include <hip/hip_runtime.h>

typedef __attribute__((ext_vector_type(16))) _Float16 v16h;
typedef __attribute__((ext_vector_type(8)))  _Float16 v8h;
typedef __attribute__((ext_vector_type(8)))  float    v8f;

#define NN   2048
#define CC   16
#define HH   32
#define WW   32
#define E_EDGES 16384
#define FEAT_PER_NODE (CC*HH*WW)   // 16384 floats per node

// ---------------- Kernel 1: signed scatter-add pooling ----------------
// pooled_pos[tgt] += feats[src] (sign>0); pooled_neg[tgt] += feats[src] (sign<0)
__global__ void scatter_pool(const float* __restrict__ feats,
                             const int*   __restrict__ edges,
                             float* __restrict__ pooled_pos,
                             float* __restrict__ pooled_neg) {
  int e    = blockIdx.x;
  int src  = edges[e*3 + 0];
  int sign = edges[e*3 + 1];
  int tgt  = edges[e*3 + 2];
  src = min(max(src, 0), NN - 1);
  tgt = min(max(tgt, 0), NN - 1);
  float* dbase = (sign > 0) ? pooled_pos : pooled_neg;
  const float* s = feats + (size_t)src * FEAT_PER_NODE;
  float*       d = dbase + (size_t)tgt * FEAT_PER_NODE;
  for (int i = threadIdx.x; i < FEAT_PER_NODE; i += blockDim.x) {
    __hip_atomic_fetch_add(&d[i], s[i], __ATOMIC_RELAXED, __HIP_MEMORY_SCOPE_AGENT);
  }
}

// ---------------- Conv1: 48 -> 32 channels, 3x3, implicit GEMM ----------------
// K = 48*9 = 432, padded to 448 (14 chunks of 32). M tile = 64 pixels (2 rows).
#define K1      448
#define K1_REAL 432
#define AS1     456   // LDS row stride (f16) for A, 912B -> 16B aligned, bank-spread
#define BS1     456

__global__ void conv1_wmma(const float* __restrict__ feats,
                           const float* __restrict__ ppos,
                           const float* __restrict__ pneg,
                           const float* __restrict__ W1,
                           const float* __restrict__ b1,
                           _Float16*    __restrict__ hout) {
  extern __shared__ char smem[];
  _Float16* Al = (_Float16*)smem;                       // 64  x AS1
  _Float16* Bl = (_Float16*)(smem + (size_t)64*AS1*2);  // 32  x BS1

  const int n   = blockIdx.x >> 4;    // image
  const int rb  = blockIdx.x & 15;    // 2-row block
  const int y0  = rb * 2;
  const int tid = threadIdx.x;

  // im2col A tile: 64 pixels x 448 (k = tap*48 + ic), f32 -> f16
  for (int e = tid; e < 64 * K1; e += 256) {
    int pix = e / K1;
    int k   = e - pix * K1;
    _Float16 v = (_Float16)0.f;
    if (k < K1_REAL) {
      int tap = k / 48;
      int ic  = k - tap * 48;
      int ky  = tap / 3, kx = tap - ky * 3;
      int y   = y0 + (pix >> 5) + ky - 1;
      int x   = (pix & 31) + kx - 1;
      if ((unsigned)y < HH && (unsigned)x < WW) {
        const float* srcp = (ic < 16) ? feats : (ic < 32) ? ppos : pneg;
        int c = ic & 15;
        v = (_Float16)srcp[(((size_t)n * CC + c) * HH + y) * WW + x];
      }
    }
    Al[pix * AS1 + k] = v;
  }
  // B tile: Bl[o][k], W1 is (32, 48, 3, 3) => idx = o*432 + ic*9 + tap
  for (int e = tid; e < 32 * K1; e += 256) {
    int o = e / K1;
    int k = e - o * K1;
    _Float16 v = (_Float16)0.f;
    if (k < K1_REAL) {
      int tap = k / 48;
      int ic  = k - tap * 48;
      v = (_Float16)W1[(size_t)o * 432 + ic * 9 + tap];
    }
    Bl[o * BS1 + k] = v;
  }
  __syncthreads();

  const int wave = tid >> 5;
  const int lane = tid & 31;
  const int pg   = wave >> 1;        // pixel group 0..3 (16 pixels each)
  const int og   = wave & 1;         // out-channel group (16 each)
  const int hi   = lane >> 4;
  const int l16  = lane & 15;

  const _Float16* arow = Al + (size_t)(pg * 16 + l16) * AS1;
  const _Float16* bcol = Bl + (size_t)(og * 16 + l16) * BS1;

  v8f acc = {};
  for (int kb = 0; kb < K1; kb += 32) {
    union { v16h v; v8h h[2]; } A, B;
    // A 16x32 f16 layout: lanes 0-15 -> K {0..7,16..23}; lanes 16-31 -> K {8..15,24..31}
    A.h[0] = *(const v8h*)(arow + kb + hi * 8);
    A.h[1] = *(const v8h*)(arow + kb + 16 + hi * 8);
    // B 32x16 f16 layout: lane = column; lanes 0-15 -> K {0..7,16..23}; lanes 16-31 -> K {8..15,24..31}
    B.h[0] = *(const v8h*)(bcol + kb + hi * 8);
    B.h[1] = *(const v8h*)(bcol + kb + 16 + hi * 8);
    acc = __builtin_amdgcn_wmma_f32_16x16x32_f16(false, A.v, false, B.v,
                                                 (short)0, acc, false, false);
  }

  const int oc   = og * 16 + l16;
  const float bias = b1[oc];
#pragma unroll
  for (int r = 0; r < 8; ++r) {
    int m = pg * 16 + (hi ? 8 + r : r);   // pixel index in 64-pixel tile
    int y = y0 + (m >> 5);
    int x = m & 31;
    float v = acc[r] + bias;
    v = v > 0.f ? v : 0.1f * v;
    hout[(((size_t)n * 32 + oc) * HH + y) * WW + x] = (_Float16)v;
  }
}

// ---------------- Conv2: 32 -> 16 channels, 3x3 ----------------
// K = 32*9 = 288 exactly (9 chunks). M tile = 128 pixels (4 rows).
#define K2  288
#define AS2 296
#define BS2 296

__global__ void conv2_wmma(const _Float16* __restrict__ hin,
                           const float* __restrict__ W2,
                           const float* __restrict__ b2,
                           float* __restrict__ out) {
  extern __shared__ char smem[];
  _Float16* Al = (_Float16*)smem;                        // 128 x AS2
  _Float16* Bl = (_Float16*)(smem + (size_t)128*AS2*2);  // 16  x BS2

  const int n   = blockIdx.x >> 3;
  const int rb  = blockIdx.x & 7;
  const int y0  = rb * 4;
  const int tid = threadIdx.x;

  for (int e = tid; e < 128 * K2; e += 256) {
    int pix = e / K2;
    int k   = e - pix * K2;
    int tap = k / 32;
    int ic  = k & 31;
    int ky  = tap / 3, kx = tap - ky * 3;
    int y   = y0 + (pix >> 5) + ky - 1;
    int x   = (pix & 31) + kx - 1;
    _Float16 v = (_Float16)0.f;
    if ((unsigned)y < HH && (unsigned)x < WW)
      v = hin[(((size_t)n * 32 + ic) * HH + y) * WW + x];
    Al[pix * AS2 + k] = v;
  }
  // W2 is (16, 32, 3, 3) => idx = o*288 + ic*9 + tap
  for (int e = tid; e < 16 * K2; e += 256) {
    int o  = e / K2;
    int k  = e - o * K2;
    int tap = k / 32;
    int ic  = k & 31;
    Bl[o * BS2 + k] = (_Float16)W2[(size_t)o * 288 + ic * 9 + tap];
  }
  __syncthreads();

  const int wave = tid >> 5;   // 0..7 -> pixel group
  const int lane = tid & 31;
  const int hi   = lane >> 4;
  const int l16  = lane & 15;

  const _Float16* arow = Al + (size_t)(wave * 16 + l16) * AS2;
  const _Float16* bcol = Bl + (size_t)l16 * BS2;

  v8f acc = {};
  for (int kb = 0; kb < K2; kb += 32) {
    union { v16h v; v8h h[2]; } A, B;
    A.h[0] = *(const v8h*)(arow + kb + hi * 8);
    A.h[1] = *(const v8h*)(arow + kb + 16 + hi * 8);
    B.h[0] = *(const v8h*)(bcol + kb + hi * 8);
    B.h[1] = *(const v8h*)(bcol + kb + 16 + hi * 8);
    acc = __builtin_amdgcn_wmma_f32_16x16x32_f16(false, A.v, false, B.v,
                                                 (short)0, acc, false, false);
  }

  const int oc   = l16;
  const float bias = b2[oc];
#pragma unroll
  for (int r = 0; r < 8; ++r) {
    int m = wave * 16 + (hi ? 8 + r : r);
    int y = y0 + (m >> 5);
    int x = m & 31;
    float v = acc[r] + bias;
    v = v > 0.f ? v : 0.1f * v;
    out[(((size_t)n * CC + oc) * HH + y) * WW + x] = v;
  }
}

// ---------------- Launch ----------------
extern "C" void kernel_launch(void* const* d_in, const int* in_sizes, int n_in,
                              void* d_out, int out_size, void* d_ws, size_t ws_size,
                              hipStream_t stream) {
  (void)in_sizes; (void)n_in; (void)out_size; (void)ws_size;
  const float* feats = (const float*)d_in[0];
  const int*   edges = (const int*)d_in[1];
  const float* W1    = (const float*)d_in[2];
  const float* b1    = (const float*)d_in[3];
  const float* W2    = (const float*)d_in[4];
  const float* b2    = (const float*)d_in[5];
  float* out = (float*)d_out;

  // Workspace layout: pooled_pos (f32, 128MB) | pooled_neg (f32, 128MB) | h (f16, 128MB)
  const size_t poolBytes = (size_t)NN * FEAT_PER_NODE * sizeof(float);
  float*    ppos = (float*)d_ws;
  float*    pneg = (float*)((char*)d_ws + poolBytes);
  _Float16* hbuf = (_Float16*)((char*)d_ws + 2 * poolBytes);

  hipMemsetAsync(d_ws, 0, 2 * poolBytes, stream);
  scatter_pool<<<E_EDGES, 256, 0, stream>>>(feats, edges, ppos, pneg);

  const size_t lds1 = (size_t)(64 * AS1 + 32 * BS1) * sizeof(_Float16);   // ~87.5 KB
  conv1_wmma<<<NN * 16, 256, lds1, stream>>>(feats, ppos, pneg, W1, b1, hbuf);

  const size_t lds2 = (size_t)(128 * AS2 + 16 * BS2) * sizeof(_Float16); // ~85.2 KB
  conv2_wmma<<<NN * 8, 256, lds2, stream>>>(hbuf, W2, b2, out);
}
